// FlaxCrossFrameAttention_76184129896667
// MI455X (gfx1250) — matrix operand order repaired and checked by reading, not compiled, not run on hardware
//
#include <hip/hip_runtime.h>

typedef __attribute__((ext_vector_type(16))) _Float16 v16h;
typedef __attribute__((ext_vector_type(8)))  float    v8f;
typedef __attribute__((ext_vector_type(4)))  int      v4i;

#if defined(__HIP_DEVICE_COMPILE__) && defined(__gfx1250__) && \
    __has_builtin(__builtin_amdgcn_global_load_async_to_lds_b128)
#define HAVE_ASYNC_LDS 1
#else
#define HAVE_ASYNC_LDS 0
#endif

union HFrag { v16h h; unsigned u[8]; };

// A fragment (16x32 f16): lane row = lane&15; K pattern {kb..kb+7, 16+kb..16+kb+7}, kb=(lane>>4)*8
__device__ __forceinline__ v16h load_a_frag(const unsigned short* base, int pitch, int lane) {
  const int row = lane & 15;
  const int kb  = (lane >> 4) << 3;
  const unsigned short* p = base + row * pitch;
  HFrag f;
#pragma unroll
  for (int i = 0; i < 4; ++i) {
    f.u[i]     = *(const unsigned*)(p + kb + 2 * i);
    f.u[i + 4] = *(const unsigned*)(p + 16 + kb + 2 * i);
  }
  return f.h;
}

// B fragment (32x16 f16) from K-transposed LDS tile ldsT[col][k]:
// lane col = colBase + (lane&15); K pattern {kb..kb+7, 16+kb..16+kb+7}, kb=(lane>>4)*8
__device__ __forceinline__ v16h load_b_frag(const unsigned short* baseT, int pitch, int lane, int colBase) {
  const int col = colBase + (lane & 15);
  const int kb  = (lane >> 4) << 3;
  const unsigned short* p = baseT + col * pitch + kb;
  HFrag f;
#pragma unroll
  for (int i = 0; i < 4; ++i) {
    f.u[i]     = *(const unsigned*)(p + 2 * i);
    f.u[i + 4] = *(const unsigned*)(p + 16 + 2 * i);
  }
  return f.h;
}

__device__ __forceinline__ v8f wmma_f16(v16h a, v16h b, v8f c) {
  return __builtin_amdgcn_wmma_f32_16x16x32_f16(false, a, false, b, (short)0, c, false, false);
}

// 16-byte global -> LDS copy; async (ASYNCcnt path) when available
__device__ __forceinline__ void copy_b128_to_lds(void* lds, const void* gptr) {
#if HAVE_ASYNC_LDS
  typedef __attribute__((address_space(1))) v4i GV;  // global int4
  typedef __attribute__((address_space(3))) v4i LV;  // LDS int4
  __builtin_amdgcn_global_load_async_to_lds_b128((GV*)gptr, (LV*)lds, 0, 0);
#else
  *(uint4*)lds = *(const uint4*)gptr;
#endif
}

__device__ __forceinline__ void wait_async0() {
#if HAVE_ASYNC_LDS
#if __has_builtin(__builtin_amdgcn_s_wait_asynccnt)
  __builtin_amdgcn_s_wait_asynccnt(0);
#else
  asm volatile("s_wait_asynccnt 0x0" ::: "memory");
#endif
#endif
}

// ---------------------------------------------------------------- utilities
__global__ void cvt_f32_f16_kernel(const float* __restrict__ in, _Float16* __restrict__ out, int n) {
  int i = blockIdx.x * blockDim.x + threadIdx.x;
  if (i < n) out[i] = (_Float16)in[i];
}

__global__ void zero_u32_kernel(unsigned* __restrict__ p, int n) {
  int i = blockIdx.x * blockDim.x + threadIdx.x;
  if (i < n) p[i] = 0u;
}

// ---------------------------------------------------------------- 128x64 GEMM
// C[m0:m0+128, n0:n0+64] = A' x B, A' row m -> A row ((m>>10)*aRowBlockStride + (m&1023))
// mode 0: f16 out, head-split: col c -> head c/80, j=c%80 ; out[((m>>10)*8+hh)*1024 + s][96]
// mode 1: f32 out = acc + bias[c], row-major [M,N]
__global__ __launch_bounds__(128) void gemm128x64_kernel(
    const _Float16* __restrict__ A, const _Float16* __restrict__ B,
    int K, int N, int aRowBlockStride, int mode,
    _Float16* __restrict__ outH, float* __restrict__ outF,
    const float* __restrict__ bias) {
  __shared__ unsigned short As[128][34];  // padded pitch: conflict-free, 4B-aligned rows
  __shared__ unsigned short BsT[64][34];  // B stored transposed: [col][k]

  const int tid = threadIdx.x;
  const int w   = tid >> 5, lane = tid & 31;
  const int hi  = lane >> 4, lo = lane & 15;
  const int n0  = blockIdx.x * 64;
  const int m0  = blockIdx.y * 128;

  v8f acc[2][4] = {};
  uint4 aReg[4], bReg[2];

  auto loadT = [&](int k0) {  // issue global loads for one K-step tile into registers
#pragma unroll
    for (int it = 0; it < 4; ++it) {
      int row = (tid >> 2) + it * 32;
      int m   = m0 + row;
      int ar  = ((m >> 10) * aRowBlockStride) + (m & 1023);
      aReg[it] = *(const uint4*)(A + (size_t)ar * K + k0 + (tid & 3) * 8);
    }
#pragma unroll
    for (int it = 0; it < 2; ++it) {
      int kr = (tid >> 3) + it * 16;
      bReg[it] = *(const uint4*)(B + (size_t)(k0 + kr) * N + n0 + (tid & 7) * 8);
    }
  };
  auto storeT = [&]() {  // registers -> LDS (B transposed)
#pragma unroll
    for (int it = 0; it < 4; ++it) {
      int row = (tid >> 2) + it * 32;
      unsigned* dst = (unsigned*)&As[row][(tid & 3) * 8];
      dst[0] = aReg[it].x; dst[1] = aReg[it].y; dst[2] = aReg[it].z; dst[3] = aReg[it].w;
    }
#pragma unroll
    for (int it = 0; it < 2; ++it) {
      int kr = (tid >> 3) + it * 16;
      int cg = (tid & 7) * 8;
      union { uint4 v; unsigned short s[8]; } u;
      u.v = bReg[it];
#pragma unroll
      for (int j = 0; j < 8; ++j) BsT[cg + j][kr] = u.s[j];
    }
  };

  loadT(0);
  storeT();
  __syncthreads();

  for (int k0 = 0; k0 < K; k0 += 32) {
    const bool more = (k0 + 32) < K;
    if (more) loadT(k0 + 32);  // global loads in flight during WMMAs

    v16h a0 = load_a_frag(&As[w * 32][0], 34, lane);
    v16h a1 = load_a_frag(&As[w * 32 + 16][0], 34, lane);
#pragma unroll
    for (int nn = 0; nn < 4; ++nn) {
      v16h b = load_b_frag(&BsT[0][0], 34, lane, nn * 16);
      acc[0][nn] = wmma_f16(a0, b, acc[0][nn]);
      acc[1][nn] = wmma_f16(a1, b, acc[1][nn]);
    }
    __syncthreads();
    if (more) {
      storeT();
      __syncthreads();
    }
  }

#pragma unroll
  for (int sub = 0; sub < 2; ++sub)
#pragma unroll
    for (int nn = 0; nn < 4; ++nn) {
      int c = n0 + nn * 16 + lo;
#pragma unroll
      for (int r = 0; r < 8; ++r) {
        int m = m0 + w * 32 + sub * 16 + hi * 8 + r;
        float v = acc[sub][nn][r];
        if (mode == 0) {
          int hh = c / 80;
          int j  = c - hh * 80;
          size_t idx = ((size_t)((m >> 10) * 8 + hh) * 1024 + (m & 1023)) * 96 + j;
          outH[idx] = (_Float16)v;
        } else {
          outF[(size_t)m * N + c] = v + bias[c];
        }
      }
    }
}

// ---------------------------------------------------------------- flash attention
// grid.x = bh (frame*8+head, 128), grid.y = 64-row Q block (16). 4 waves, 16 Q rows each.
// K/V from group-first frame: kvbh = (frame>>3)*8 + head. Head dim 80 padded to 96.
// Double-buffered K/V LDS tiles; K staged via async global->LDS copies.
__global__ __launch_bounds__(128) void FlaxCrossFrameAttention_flash_kernel(
    const _Float16* __restrict__ Qp, const _Float16* __restrict__ Kp,
    const _Float16* __restrict__ Vp, _Float16* __restrict__ Oh) {
  __shared__ _Float16 Ks[2][32][98];    // K rows [key][kd]  (B-frag reads consecutive in kd)
  __shared__ _Float16 VsT[2][96][34];   // V transposed [dcol][key]
  __shared__ _Float16 Ps[4][16][34];    // per-wave P tile [qrow][key]

  const int tid  = threadIdx.x;
  const int w    = tid >> 5, lane = tid & 31;
  const int hi   = lane >> 4, lo = lane & 15;
  const int bh   = blockIdx.x;
  const int f    = bh >> 3, h = bh & 7;
  const int kvbh = (f >> 3) * 8 + h;
  const int qr0  = blockIdx.y * 64 + w * 16;

  const _Float16* Kbase = Kp + (size_t)kvbh * 1024 * 96;
  const _Float16* Vbase = Vp + (size_t)kvbh * 1024 * 96;

  auto stageK = [&](int kb0, int buf) {
#pragma unroll
    for (int it = 0; it < 3; ++it) {
      int slot = tid + it * 128;
      int key  = slot / 12;
      int cg   = (slot - key * 12) * 8;
      copy_b128_to_lds(&Ks[buf][key][cg], Kbase + (size_t)(kb0 + key) * 96 + cg);
    }
  };
  auto stageV = [&](int kb0, int buf) {
#pragma unroll
    for (int it = 0; it < 3; ++it) {
      int slot = tid + it * 128;
      int key  = slot / 12;
      int cg   = (slot - key * 12) * 8;
      union { uint4 v; _Float16 s[8]; } u;
      u.v = *(const uint4*)(Vbase + (size_t)(kb0 + key) * 96 + cg);
#pragma unroll
      for (int j = 0; j < 8; ++j) VsT[buf][cg + j][key] = u.s[j];
    }
  };

  // preload Q fragments for this wave's 16 rows (3 chunks of K=32 over padded dim 96)
  v16h qa[3];
  {
    const unsigned short* qptr =
        (const unsigned short*)(Qp + ((size_t)bh * 1024 + qr0 + lo) * 96);
    const int kb = hi << 3;
#pragma unroll
    for (int kk = 0; kk < 3; ++kk) {
      HFrag fg;
#pragma unroll
      for (int i = 0; i < 4; ++i) {
        fg.u[i]     = *(const unsigned*)(qptr + kk * 32 + kb + 2 * i);
        fg.u[i + 4] = *(const unsigned*)(qptr + kk * 32 + 16 + kb + 2 * i);
      }
      qa[kk] = fg.h;
    }
  }

  v8f o[5] = {};
  float mrow[8], lsum[8], alpha[8];
#pragma unroll
  for (int r = 0; r < 8; ++r) { mrow[r] = -1e30f; lsum[r] = 0.f; }

  const float scl2 = 0.11180339887f * 1.44269504089f;  // 80^-0.5 * log2(e)

  stageK(0, 0);
  stageV(0, 0);
  wait_async0();
  __syncthreads();

  for (int it = 0; it < 32; ++it) {
    const int kb0 = it * 32;
    const int cur = it & 1;
    if (it < 31) {  // prefetch next K/V block into the other buffer (overlaps compute)
      stageK(kb0 + 32, cur ^ 1);
      stageV(kb0 + 32, cur ^ 1);
    }

    // S = Q K^T : two 16x16 score tiles over 32 keys, 3 k-chunks each
    v8f s0 = {}, s1 = {};
#pragma unroll
    for (int kk = 0; kk < 3; ++kk) {
      v16h b0 = load_b_frag((const unsigned short*)&Ks[cur][0][kk * 32], 98, lane, 0);
      v16h b1 = load_b_frag((const unsigned short*)&Ks[cur][0][kk * 32], 98, lane, 16);
      s0 = wmma_f16(qa[kk], b0, s0);
      s1 = wmma_f16(qa[kk], b1, s1);
    }

    // online softmax: rows live as (r, lane-half); reduce across 16-lane groups
#pragma unroll
    for (int r = 0; r < 8; ++r) {
      float a0 = s0[r] * scl2;
      float a1 = s1[r] * scl2;
      float t  = fmaxf(a0, a1);
#pragma unroll
      for (int msk = 1; msk < 16; msk <<= 1) t = fmaxf(t, __shfl_xor(t, msk, 32));
      float mn = fmaxf(mrow[r], t);
      float al = exp2f(mrow[r] - mn);
      float p0 = exp2f(a0 - mn);
      float p1 = exp2f(a1 - mn);
      float ps = p0 + p1;
#pragma unroll
      for (int msk = 1; msk < 16; msk <<= 1) ps += __shfl_xor(ps, msk, 32);
      lsum[r]  = lsum[r] * al + ps;
      mrow[r]  = mn;
      alpha[r] = al;
      s0[r] = p0; s1[r] = p1;
    }
#pragma unroll
    for (int j = 0; j < 5; ++j)
#pragma unroll
      for (int r = 0; r < 8; ++r) o[j][r] *= alpha[r];

    // P: C-layout -> LDS -> A-layout for P*V
#pragma unroll
    for (int r = 0; r < 8; ++r) {
      int row = hi * 8 + r;
      Ps[w][row][lo]      = (_Float16)s0[r];
      Ps[w][row][16 + lo] = (_Float16)s1[r];
    }
    __syncthreads();

    v16h pa = load_a_frag((const unsigned short*)&Ps[w][0][0], 34, lane);
#pragma unroll
    for (int j = 0; j < 5; ++j) {
      v16h b = load_b_frag((const unsigned short*)&VsT[cur][0][0], 34, lane, j * 16);
      o[j] = wmma_f16(pa, b, o[j]);
    }

    wait_async0();    // this wave's async K copy for buffer cur^1 done
    __syncthreads();  // all waves done reading buffer cur and done staging cur^1
  }

  // normalize and write O as f16 [F, S, 640] (head h occupies cols h*80..h*80+79)
#pragma unroll
  for (int r = 0; r < 8; ++r) lsum[r] = 1.f / lsum[r];
  const int srow0 = qr0 + hi * 8;
#pragma unroll
  for (int j = 0; j < 5; ++j) {
    int dcol = j * 16 + lo;
#pragma unroll
    for (int r = 0; r < 8; ++r) {
      size_t idx = ((size_t)f * 1024 + (srow0 + r)) * 640 + h * 80 + dcol;
      Oh[idx] = (_Float16)(o[j][r] * lsum[r]);
    }
  }
}

// ---------------------------------------------------------------- launch
extern "C" void kernel_launch(void* const* d_in, const int* in_sizes, int n_in,
                              void* d_out, int out_size, void* d_ws, size_t ws_size,
                              hipStream_t stream) {
  const float* X  = (const float*)d_in[0];
  const float* Wq = (const float*)d_in[1];
  const float* Wk = (const float*)d_in[2];
  const float* Wv = (const float*)d_in[3];
  const float* Wo = (const float*)d_in[4];
  const float* bo = (const float*)d_in[5];
  float* out = (float*)d_out;

  char* ws = (char*)d_ws;
  _Float16* Xh  = (_Float16*)(ws + 0);          // 16384x640
  _Float16* Wqh = (_Float16*)(ws + 20971520);   // 640x640
  _Float16* Wkh = (_Float16*)(ws + 21790720);
  _Float16* Wvh = (_Float16*)(ws + 22609920);
  _Float16* Woh = (_Float16*)(ws + 23429120);
  _Float16* Qp  = (_Float16*)(ws + 24248320);   // 128x1024x96 (padded heads)
  _Float16* Kp  = (_Float16*)(ws + 49414144);   // 16x1024x96
  _Float16* Vp  = (_Float16*)(ws + 52559872);   // 16x1024x96
  _Float16* Oh  = (_Float16*)(ws + 55705600);   // 16384x640

  // f32 -> f16 conversions
  cvt_f32_f16_kernel<<<40960, 256, 0, stream>>>(X,  Xh,  16384 * 640);
  cvt_f32_f16_kernel<<<1600,  256, 0, stream>>>(Wq, Wqh, 640 * 640);
  cvt_f32_f16_kernel<<<1600,  256, 0, stream>>>(Wk, Wkh, 640 * 640);
  cvt_f32_f16_kernel<<<1600,  256, 0, stream>>>(Wv, Wvh, 640 * 640);
  cvt_f32_f16_kernel<<<1600,  256, 0, stream>>>(Wo, Woh, 640 * 640);

  // zero Qp/Kp/Vp (contiguous) so head-dim pad columns 80..95 are zero
  zero_u32_kernel<<<30720, 256, 0, stream>>>((unsigned*)Qp, 7864320);

  // projections: Q over all 16 frames; K/V only over group-first frames (0 and 8)
  gemm128x64_kernel<<<dim3(10, 128), 128, 0, stream>>>(Xh, Wqh, 640, 640, 1024, 0, Qp, nullptr, nullptr);
  gemm128x64_kernel<<<dim3(10, 16),  128, 0, stream>>>(Xh, Wkh, 640, 640, 8192, 0, Kp, nullptr, nullptr);
  gemm128x64_kernel<<<dim3(10, 16),  128, 0, stream>>>(Xh, Wvh, 640, 640, 8192, 0, Vp, nullptr, nullptr);

  // flash attention: 128 (frame,head) problems x 16 Q-blocks
  FlaxCrossFrameAttention_flash_kernel<<<dim3(128, 16), 128, 0, stream>>>(Qp, Kp, Vp, Oh);

  // output projection + bias -> f32
  gemm128x64_kernel<<<dim3(10, 128), 128, 0, stream>>>(Oh, Woh, 640, 640, 1024, 1, nullptr, out, bo);
}
